// MultiBiLSTMLayer_83726092468475
// MI455X (gfx1250) — matrix operand order repaired and checked by reading, not compiled, not run on hardware
//
#include <hip/hip_runtime.h>

// ---------------------------------------------------------------------------
// MultiBiLSTMLayer for MI455X (gfx1250): bf16 WMMA GEMMs + persistent-WG scan
// + Tensor Data Mover (TDM) staging of B panels / pre tiles into LDS.
// B=64, T=1024, D=128, H=256, 4H=1024.  wave32, 16x16x32 bf16 WMMA.
// ---------------------------------------------------------------------------

typedef __attribute__((ext_vector_type(16))) __bf16        bf16x16;
typedef __attribute__((ext_vector_type(8)))  float         f32x8;
typedef __attribute__((ext_vector_type(4)))  unsigned int  u32x4;
typedef __attribute__((ext_vector_type(8)))  int           i32x8;
typedef __attribute__((ext_vector_type(4)))  int           i32x4;

union FragU { uint4 q[2]; bf16x16 v; };

static constexpr int Bsz = 64;
static constexpr int Tsz = 1024;
static constexpr int Hsz = 256;
static constexpr int Gsz = 1024;   // 4H
static constexpr int ZP  = 1032;   // padded z row stride (floats)
static constexpr int HP  = 264;    // padded h row stride (bf16)
// scan LDS: zbuf + cbuf + hbuf + double-buffered pre tile [2][16][1024] f32
static constexpr size_t SCAN_LDS =
    (size_t)16*ZP*4 + (size_t)16*256*4 + (size_t)16*HP*2 + (size_t)2*16*1024*4;

__device__ __forceinline__ unsigned short f2bf(float f) {
  unsigned int u = __float_as_uint(f);
  u += 0x7fffu + ((u >> 16) & 1u);          // round-to-nearest-even
  return (unsigned short)(u >> 16);
}
__device__ __forceinline__ float sigf(float x)   { return 1.0f / (1.0f + __expf(-x)); }
__device__ __forceinline__ float tanh_f(float x) { return 2.0f * sigf(2.0f * x) - 1.0f; }

// generic LDS pointer -> LDS byte offset (aperture decode keeps offset in addr[31:0])
__device__ __forceinline__ unsigned lds_off_of(const void* p) {
  return (unsigned)(unsigned long long)p;
}

// ------------------------ Tensor Data Mover (TDM) --------------------------
// 2-D tile load Global->LDS.  D# packed per CDNA5 ISA §8.3/8.4.
// data_size_code: 0=1B,1=2B,2=4B,3=8B.  pad: +(pad_amount+1) DWORDs every
// (2<<pad_interval) DWORDs stored (pad_enable=0 => disabled).
__device__ __forceinline__ void tdm_load_2d(
    unsigned lds_off, const void* gaddr,
    unsigned tile_d0, unsigned tile_d1,
    unsigned tensor_d0, unsigned tensor_d1,
    unsigned long long d0_stride,            // elements between dim1 lines
    unsigned data_size_code,
    unsigned pad_enable, unsigned pad_interval, unsigned pad_amount) {
  unsigned long long ga = (unsigned long long)gaddr;
  u32x4 g0;
  g0[0] = 1u;                                              // count=1, user mode
  g0[1] = lds_off;                                         // lds_addr
  g0[2] = (unsigned)(ga & 0xffffffffu);                    // global_addr[31:0]
  g0[3] = (unsigned)((ga >> 32) & 0x01ffffffu) | (2u << 30); // addr[56:32] | type=2
  i32x8 g1;
  g1[0] = (int)((data_size_code << 16) | (pad_enable << 20) |
                (pad_interval << 22) | (pad_amount << 25));
  g1[1] = (int)((tensor_d0 & 0xffffu) << 16);              // tensor_dim0[15:0]
  g1[2] = (int)(((tensor_d0 >> 16) & 0xffffu) | ((tensor_d1 & 0xffffu) << 16));
  g1[3] = (int)(((tensor_d1 >> 16) & 0xffffu) | ((tile_d0 & 0xffffu) << 16));
  g1[4] = (int)(tile_d1 & 0xffffu);                        // tile_dim2 = 0
  g1[5] = (int)(unsigned)(d0_stride & 0xffffffffu);
  g1[6] = (int)(unsigned)((d0_stride >> 32) & 0xffffu);    // dim1_stride = 0
  g1[7] = 0;
  i32x4 z4 = {0, 0, 0, 0};
  i32x8 z8 = {0, 0, 0, 0, 0, 0, 0, 0};
  // 6-arg toolchain variant: (g0, g1, grp2, grp3, grp4, cpol)
  __builtin_amdgcn_tensor_load_to_lds(g0, g1, z4, z4, z8, 0);
}

// A frag (16-bit 16x32): lane half=0 -> K kb..kb+15; half=1 -> +16
__device__ __forceinline__ bf16x16 load_a(const unsigned short* row, int half, int kb) {
  FragU f;
  f.q[0] = *(const uint4*)(row + kb + half * 16);
  f.q[1] = *(const uint4*)(row + kb + half * 16 + 8);
  return f.v;
}
// B frag (16-bit 32x16, N-major packed): lane half=0 -> K kb..kb+15; half=1 -> +16
__device__ __forceinline__ bf16x16 load_b(const unsigned short* row, int half, int kb) {
  FragU f;
  f.q[0] = *(const uint4*)(row + kb + half * 16);
  f.q[1] = *(const uint4*)(row + kb + half * 16 + 8);
  return f.v;
}

// --------------------------- prep kernels ----------------------------------

// out[n*K + k] = bf16(in[k*1024 + n])   (transpose W/U to N-major for B frags)
__global__ __launch_bounds__(256) void pack_bf16_t_kernel(
    const float* __restrict__ in, unsigned short* __restrict__ out, int K) {
  int idx = blockIdx.x * 256 + threadIdx.x;
  if (idx >= K * 1024) return;
  int n = idx / K, k = idx - n * K;
  out[idx] = f2bf(in[(size_t)k * 1024 + n]);
}

// xb[(t*64+b)*128 + k] = bf16(x[(b*1024+t)*128 + k])
__global__ __launch_bounds__(256) void convert_x_kernel(
    const float* __restrict__ x, unsigned short* __restrict__ xb) {
  int idx = blockIdx.x * 256 + threadIdx.x;
  if (idx >= Tsz * Bsz * 128) return;
  int k  = idx & 127;
  int tb = idx >> 7;
  int b  = tb & 63, t = tb >> 6;
  xb[idx] = f2bf(x[((size_t)b * Tsz + t) * 128 + k]);
}

// ----------------------- input-projection GEMM -----------------------------
// out[M,1024] = A[M,K](bf16) @ Bp[1024,K]^T(bf16, N-major) + bias ; M = T*B.
// TDM stages the block's [64 x K] B panel into LDS once (pad feature inserts
// +8 halfwords/row for bank spreading); all 8 waves reuse it via ds_load_b128.
__global__ __launch_bounds__(256) void gemm_pre_kernel(
    const unsigned short* __restrict__ A, const unsigned short* __restrict__ Bp,
    const float* __restrict__ bias, float* __restrict__ out,
    int K, int KP, int padI) {
  extern __shared__ unsigned short bpanel[];     // [64][KP] bf16
  const int tid = threadIdx.x, wave = tid >> 5, lane = tid & 31;
  const int half = lane >> 4, lm = lane & 15;
  const int mbase = blockIdx.x * 128 + wave * 16;
  const int nbase = blockIdx.y * 64;

  if (wave == 0) {
    tdm_load_2d(lds_off_of(bpanel), Bp + (size_t)nbase * K,
                /*tile*/ K, 64, /*tensor*/ K, 64, /*stride*/ (unsigned long long)K,
                /*2B*/ 1, /*pad*/ 1, (unsigned)padI, 3);   // +4 DWORDs per row
    __builtin_amdgcn_s_wait_tensorcnt(0);
  }
  __syncthreads();

  f32x8 acc[4];
  #pragma unroll
  for (int ti = 0; ti < 4; ++ti) {
    float bv = bias[nbase + ti * 16 + lm];
    #pragma unroll
    for (int r = 0; r < 8; ++r) acc[ti][r] = bv;
  }
  const unsigned short* arow = A + (size_t)(mbase + lm) * K;
  #pragma unroll 4
  for (int kb = 0; kb < K; kb += 32) {
    bf16x16 a = load_a(arow, half, kb);
    #pragma unroll
    for (int ti = 0; ti < 4; ++ti) {
      bf16x16 b = load_b(bpanel + (size_t)(ti * 16 + lm) * KP, half, kb);
      acc[ti] = __builtin_amdgcn_wmma_f32_16x16x32_bf16(
          false, a, false, b, (short)0, acc[ti], false, false);
    }
  }
  #pragma unroll
  for (int ti = 0; ti < 4; ++ti) {
    int n = nbase + ti * 16 + lm;
    #pragma unroll
    for (int r = 0; r < 8; ++r) {
      int m = mbase + r + half * 8;
      out[(size_t)m * Gsz + n] = acc[ti][r];
    }
  }
}

// ----------------------------- LSTM scan -----------------------------------
// grid = 8 blocks: dir(2) x batch-chunk(4 of 16 rows). Persistent over T.
// TDM double-buffers the 64KB pre[t] tile (the only HBM reads) one step
// ahead; U streams from L2 via pipelined b128 loads into WMMA.
__global__ __launch_bounds__(256) void lstm_scan_kernel(
    const float* __restrict__ pre_f, const float* __restrict__ pre_b,
    const unsigned short* __restrict__ Upf, const unsigned short* __restrict__ Upb,
    float* __restrict__ hseq_f, float* __restrict__ hseq_b) {
  extern __shared__ char smem[];
  float*          zbuf = (float*)smem;                       // [16][ZP]
  float*          cbuf = zbuf + 16 * ZP;                     // [16][256]
  unsigned short* hbuf = (unsigned short*)(cbuf + 16 * 256); // [16][HP] bf16
  float*          pbuf = (float*)(hbuf + 16 * HP);           // [2][16][1024]

  const int dir = blockIdx.x >> 2;
  const int b0  = (blockIdx.x & 3) * 16;
  const float* pre          = dir ? pre_b  : pre_f;
  const unsigned short* U   = dir ? Upb    : Upf;
  float* hseq               = dir ? hseq_b : hseq_f;

  const int tid  = threadIdx.x;
  const int wave = tid >> 5, lane = tid & 31;
  const int half = lane >> 4, lm = lane & 15;

  for (int i = tid; i < 16 * 256; i += 256) cbuf[i] = 0.0f;
  for (int i = tid; i < 16 * HP;  i += 256) hbuf[i] = 0;

  if (wave == 0) {     // prologue: prefetch first pre tile
    int t0 = dir ? (Tsz - 1) : 0;
    tdm_load_2d(lds_off_of(pbuf), pre + ((size_t)t0 * Bsz + b0) * Gsz,
                /*tile*/ Gsz, 16, /*tensor*/ Gsz, 16,
                /*stride*/ (unsigned long long)Bsz * Gsz, /*4B*/ 2, 0, 0, 0);
  }
  __syncthreads();

  for (int step = 0; step < Tsz; ++step) {
    const int t = dir ? (Tsz - 1 - step) : step;

    if (wave == 0) {   // prefetch next tile, then wait for current
      if (step + 1 < Tsz) {
        int tn = dir ? (Tsz - 2 - step) : (step + 1);
        tdm_load_2d(lds_off_of(pbuf) + (unsigned)(((step + 1) & 1) * 16 * 1024 * 4),
                    pre + ((size_t)tn * Bsz + b0) * Gsz,
                    Gsz, 16, Gsz, 16, (unsigned long long)Bsz * Gsz, 2, 0, 0, 0);
        __builtin_amdgcn_s_wait_tensorcnt(1);
      } else {
        __builtin_amdgcn_s_wait_tensorcnt(0);
      }
    }
    __syncthreads();
    const float* prow = pbuf + (size_t)(step & 1) * 16 * 1024;

    f32x8 acc[8];
    #pragma unroll
    for (int ti = 0; ti < 8; ++ti) {            // C := pre tile (from LDS)
      int n = (wave * 8 + ti) * 16 + lm;
      #pragma unroll
      for (int r = 0; r < 8; ++r)
        acc[ti][r] = prow[(size_t)(r + half * 8) * Gsz + n];
    }
    #pragma unroll 2
    for (int k = 0; k < 8; ++k) {               // K = 256 in steps of 32
      int kb = k * 32;
      bf16x16 a = load_a(hbuf + lm * HP, half, kb);
      #pragma unroll
      for (int ti = 0; ti < 8; ++ti) {
        const unsigned short* brow = U + (size_t)((wave * 8 + ti) * 16 + lm) * Hsz;
        bf16x16 b = load_b(brow, half, kb);
        acc[ti] = __builtin_amdgcn_wmma_f32_16x16x32_bf16(
            false, a, false, b, (short)0, acc[ti], false, false);
      }
    }
    #pragma unroll
    for (int ti = 0; ti < 8; ++ti) {            // stage z
      int n = (wave * 8 + ti) * 16 + lm;
      #pragma unroll
      for (int r = 0; r < 8; ++r)
        zbuf[(r + half * 8) * ZP + n] = acc[ti][r];
    }
    __syncthreads();

    {                                           // gates + state update
      const int m = tid >> 4;
      const int j0 = tid & 15;
      float* zr = zbuf + m * ZP;
      #pragma unroll
      for (int e = 0; e < 16; ++e) {
        int j = j0 + e * 16;
        float ig = sigf(zr[j]);
        float fg = sigf(zr[256 + j]);
        float gg = tanh_f(zr[512 + j]);
        float og = sigf(zr[768 + j]);
        float c  = fg * cbuf[m * 256 + j] + ig * gg;
        cbuf[m * 256 + j] = c;
        float h = og * tanh_f(c);
        hbuf[m * HP + j] = f2bf(h);
        hseq[((size_t)t * Bsz + (b0 + m)) * Hsz + j] = h;
      }
    }
    __syncthreads();
  }
}

// ------------------------------ LayerNorm ----------------------------------
__global__ __launch_bounds__(256) void ln_kernel(
    const float* __restrict__ hf, const float* __restrict__ hb,
    const float* __restrict__ gamma, const float* __restrict__ beta,
    unsigned short* __restrict__ out_bf, float* __restrict__ out_f32) {
  int row  = blockIdx.x * 8 + (threadIdx.x >> 5);   // t*64 + b
  int lane = threadIdx.x & 31;
  int t = row >> 6, b = row & 63;
  const float* pf = hf + (size_t)row * Hsz;
  const float* pb = hb + (size_t)row * Hsz;

  float v[16], s = 0.0f;
  #pragma unroll
  for (int e = 0; e < 16; ++e) {
    int j = lane + e * 32;
    float x = (j < Hsz) ? pf[j] : pb[j - Hsz];
    v[e] = x; s += x;
  }
  #pragma unroll
  for (int off = 16; off > 0; off >>= 1) s += __shfl_xor(s, off, 32);
  float mu = s * (1.0f / 512.0f);
  float q = 0.0f;
  #pragma unroll
  for (int e = 0; e < 16; ++e) { float d = v[e] - mu; q += d * d; }
  #pragma unroll
  for (int off = 16; off > 0; off >>= 1) q += __shfl_xor(q, off, 32);
  float rstd = rsqrtf(q * (1.0f / 512.0f) + 0.001f);
  #pragma unroll
  for (int e = 0; e < 16; ++e) {
    int j = lane + e * 32;
    float y = (v[e] - mu) * rstd * gamma[j] + beta[j];
    if (out_bf)  out_bf[(size_t)row * 512 + j] = f2bf(y);
    if (out_f32) out_f32[((size_t)b * Tsz + t) * 512 + j] = y;
  }
}

// ------------------------------- host --------------------------------------

extern "C" void kernel_launch(void* const* d_in, const int* in_sizes, int n_in,
                              void* d_out, int out_size, void* d_ws, size_t ws_size,
                              hipStream_t stream) {
  (void)in_sizes; (void)n_in; (void)out_size; (void)ws_size;
  const float* x    = (const float*)d_in[0];
  const float* W1f  = (const float*)d_in[1];
  const float* U1f  = (const float*)d_in[2];
  const float* b1f  = (const float*)d_in[3];
  const float* W1b  = (const float*)d_in[4];
  const float* U1b  = (const float*)d_in[5];
  const float* b1b  = (const float*)d_in[6];
  const float* g1   = (const float*)d_in[7];
  const float* be1  = (const float*)d_in[8];
  const float* W2f  = (const float*)d_in[9];
  const float* U2f  = (const float*)d_in[10];
  const float* b2f  = (const float*)d_in[11];
  const float* W2b  = (const float*)d_in[12];
  const float* U2b  = (const float*)d_in[13];
  const float* b2b  = (const float*)d_in[14];
  const float* g2   = (const float*)d_in[15];
  const float* be2  = (const float*)d_in[16];

  char* ws = (char*)d_ws;
  size_t off = 0;
  auto take = [&](size_t bytes) -> char* {
    char* p = ws + off;
    off += (bytes + 255) & ~(size_t)255;
    return p;
  };
  const size_t MR = (size_t)Tsz * Bsz;                       // 65536 rows
  unsigned short* xb1   = (unsigned short*)take(MR * 128 * 2);
  unsigned short* wp1f  = (unsigned short*)take((size_t)1024 * 128 * 2);
  unsigned short* wp1b  = (unsigned short*)take((size_t)1024 * 128 * 2);
  unsigned short* up1f  = (unsigned short*)take((size_t)1024 * 256 * 2);
  unsigned short* up1b  = (unsigned short*)take((size_t)1024 * 256 * 2);
  unsigned short* wp2f  = (unsigned short*)take((size_t)1024 * 512 * 2);
  unsigned short* wp2b  = (unsigned short*)take((size_t)1024 * 512 * 2);
  unsigned short* up2f  = (unsigned short*)take((size_t)1024 * 256 * 2);
  unsigned short* up2b  = (unsigned short*)take((size_t)1024 * 256 * 2);
  float*          pre_f = (float*)take(MR * Gsz * 4);        // 256 MB
  float*          pre_b = (float*)take(MR * Gsz * 4);        // 256 MB
  float*          hsf   = (float*)take(MR * Hsz * 4);        // 64 MB
  float*          hsb   = (float*)take(MR * Hsz * 4);        // 64 MB
  unsigned short* h1b16 = (unsigned short*)take(MR * 512 * 2); // 64 MB

  dim3 blk(256);
  pack_bf16_t_kernel<<<(1024*128 + 255) / 256, blk, 0, stream>>>(W1f, wp1f, 128);
  pack_bf16_t_kernel<<<(1024*128 + 255) / 256, blk, 0, stream>>>(W1b, wp1b, 128);
  pack_bf16_t_kernel<<<(1024*256 + 255) / 256, blk, 0, stream>>>(U1f, up1f, 256);
  pack_bf16_t_kernel<<<(1024*256 + 255) / 256, blk, 0, stream>>>(U1b, up1b, 256);
  pack_bf16_t_kernel<<<(1024*512 + 255) / 256, blk, 0, stream>>>(W2f, wp2f, 512);
  pack_bf16_t_kernel<<<(1024*512 + 255) / 256, blk, 0, stream>>>(W2b, wp2b, 512);
  pack_bf16_t_kernel<<<(1024*256 + 255) / 256, blk, 0, stream>>>(U2f, up2f, 256);
  pack_bf16_t_kernel<<<(1024*256 + 255) / 256, blk, 0, stream>>>(U2b, up2b, 256);
  convert_x_kernel<<<(int)((MR * 128 + 255) / 256), blk, 0, stream>>>(x, xb1);

  dim3 ggrid(512, 16);  // 128x64 tiles over [65536 x 1024]
  // B-panel LDS bytes: 64 rows * (K+8) halfwords; pad interval code: 2<<c DWORDs
  const size_t lds1 = (size_t)64 * (128 + 8) * 2;   // K=128, rows of 64 DW -> code 5
  const size_t lds2 = (size_t)64 * (512 + 8) * 2;   // K=512, rows of 256 DW -> code 7

  // layer 1
  gemm_pre_kernel<<<ggrid, blk, lds1, stream>>>(xb1, wp1f, b1f, pre_f, 128, 136, 5);
  gemm_pre_kernel<<<ggrid, blk, lds1, stream>>>(xb1, wp1b, b1b, pre_b, 128, 136, 5);
  lstm_scan_kernel<<<8, blk, SCAN_LDS, stream>>>(pre_f, pre_b, up1f, up1b, hsf, hsb);
  ln_kernel<<<(int)(MR / 8), blk, 0, stream>>>(hsf, hsb, g1, be1, h1b16, nullptr);
  // layer 2
  gemm_pre_kernel<<<ggrid, blk, lds2, stream>>>(h1b16, wp2f, b2f, pre_f, 512, 520, 7);
  gemm_pre_kernel<<<ggrid, blk, lds2, stream>>>(h1b16, wp2b, b2b, pre_b, 512, 520, 7);
  lstm_scan_kernel<<<8, blk, SCAN_LDS, stream>>>(pre_f, pre_b, up2f, up2b, hsf, hsb);
  ln_kernel<<<(int)(MR / 8), blk, 0, stream>>>(hsf, hsb, g2, be2, nullptr, (float*)d_out);
}
